// WindowAttention_30657476559222
// MI455X (gfx1250) — matrix operand order repaired and checked by reading, not compiled, not run on hardware
//
#include <hip/hip_runtime.h>

typedef __attribute__((ext_vector_type(16))) _Float16 v16h;
typedef __attribute__((ext_vector_type(8)))  _Float16 v8h;
typedef __attribute__((ext_vector_type(8)))  float    v8f;

#define DIMC   256
#define NHEAD  8
#define HDIM   32
#define NTOK   64          // 8x8 window tokens
#define XS     264         // row stride (halves) for 64 x 256 row-major LDS slabs (+8 pad)
#define VTS    72          // row stride (halves) for v^T  [256][64] (+8 pad)
#define PSTR   72          // row stride (halves) for P    [64][64] per head (+8 pad)
#define SCALEF 0.17677669529663687f   // 32^-0.5

// halves: x | q | k | vT | P | out
#define LDS_HALVES (4*NTOK*XS + DIMC*VTS + NHEAD*NTOK*PSTR)

__device__ __forceinline__ v8f wmma32(v16h a, v16h b, v8f c) {
  return __builtin_amdgcn_wmma_f32_16x16x32_f16(false, a, false, b, (short)0, c,
                                                false, false);
}

// A fragment, 16x32 f16, row-major source with given stride (halves).
// ISA layout: lane m(0..15)/hi: halves 0..7 = A[m][8*hi + j], 8..15 = A[m][16 + 8*hi + j]
__device__ __forceinline__ v16h load_a16x32(const _Float16* tile, int stride) {
  const int lane = threadIdx.x & 31;
  const int m = lane & 15, hi = lane >> 4;
  const _Float16* p = tile + m * stride + hi * 8;
  v8h lo = *(const v8h*)p;
  v8h hh = *(const v8h*)(p + 16);
  return __builtin_shufflevector(lo, hh, 0, 1, 2, 3, 4, 5, 6, 7,
                                         8, 9, 10, 11, 12, 13, 14, 15);
}

// B fragment, 32x16, where B[k][n] = src[n][k] (src row-major [N][K], stride in halves).
// ISA layout: lane n(0..15)/kh: halves 0..7 = B[8*kh + j][n], 8..15 = B[16 + 8*kh + j][n]
__device__ __forceinline__ v16h load_b32x16(const _Float16* src, int stride) {
  const int lane = threadIdx.x & 31;
  const int n = lane & 15, kh = lane >> 4;
  const _Float16* p = src + n * stride + kh * 8;
  v8h lo = *(const v8h*)p;
  v8h hh = *(const v8h*)(p + 16);
  return __builtin_shufflevector(lo, hh, 0, 1, 2, 3, 4, 5, 6, 7,
                                         8, 9, 10, 11, 12, 13, 14, 15);
}

__global__ void wa_convert_weights(const float* __restrict__ wqkv,
                                   const float* __restrict__ wproj,
                                   _Float16* __restrict__ wqkv_h,
                                   _Float16* __restrict__ wproj_h) {
  int i = blockIdx.x * blockDim.x + threadIdx.x;           // 768 blocks * 256 = 196608
  wqkv_h[i] = (_Float16)wqkv[i];
  if (i < DIMC * DIMC) wproj_h[i] = (_Float16)wproj[i];
}

__global__ __launch_bounds__(256)
void wa_fused(const float* __restrict__ x,
              const float* __restrict__ bqkv,
              const float* __restrict__ bproj,
              const _Float16* __restrict__ wqkv_h,
              const _Float16* __restrict__ wproj_h,
              float* __restrict__ out) {
  extern __shared__ _Float16 smem[];
  _Float16* lx  = smem;                  // [64][XS]
  _Float16* lq  = lx  + NTOK * XS;       // [64][XS]   channels 0..255
  _Float16* lk  = lq  + NTOK * XS;       // [64][XS]
  _Float16* lvt = lk  + NTOK * XS;       // [256][VTS] v transposed: [dim][token]
  _Float16* lp  = lvt + DIMC * VTS;      // [8][64][PSTR] per-head P
  _Float16* lo  = lp  + NHEAD * NTOK * PSTR; // [64][XS] attention output

  const int win = blockIdx.x;            // 4096 windows
  const int b   = win >> 10;             // / (32*32)
  const int hw  = (win >> 5) & 31;
  const int ww  = win & 31;

  const int w    = threadIdx.x >> 5;     // wave id 0..7
  const int lane = threadIdx.x & 31;
  const int ln   = lane & 15;
  const int hi   = lane >> 4;

  // ---- Stage 1: gather window of x -> LDS f16 [token][channel] ----
  const int px_base = ((b * DIMC) * 256 + hw * 8) * 256 + ww * 8;
  for (int i = threadIdx.x; i < NTOK * DIMC; i += 256) {
    const int t = i & 63, c = i >> 6;
    const int ws1 = t >> 3, ws2 = t & 7;
    const float xv = x[px_base + (c * 256 + ws1) * 256 + ws2];
    lx[t * XS + c] = (_Float16)xv;
  }
  __syncthreads();

  // ---- Stage 2: qkv = xw @ Wqkv^T + b   (M=64, N=768, K=256) ----
  // wave w owns n-tiles [6w, 6w+6); q/k stored row-major, v stored transposed.
  for (int mt = 0; mt < 4; ++mt) {
    v16h af[8];
#pragma unroll
    for (int ks = 0; ks < 8; ++ks)
      af[ks] = load_a16x32(lx + (mt * 16) * XS + ks * 32, XS);
    for (int j = 0; j < 6; ++j) {
      const int nt = __builtin_amdgcn_readfirstlane(w * 6 + j);  // wave-uniform -> scalar branch
      v8f acc = {};
#pragma unroll
      for (int ks = 0; ks < 8; ++ks) {
        v16h bf = load_b32x16(wqkv_h + (nt * 16) * DIMC + ks * 32, DIMC);
        acc = wmma32(af[ks], bf, acc);
      }
      const int col = nt * 16 + ln;      // 0..767
      const float bias = bqkv[col];
      if (nt < 16) {                     // q
#pragma unroll
        for (int r = 0; r < 8; ++r)
          lq[(mt * 16 + hi * 8 + r) * XS + col] = (_Float16)(acc[r] + bias);
      } else if (nt < 32) {              // k
#pragma unroll
        for (int r = 0; r < 8; ++r)
          lk[(mt * 16 + hi * 8 + r) * XS + (col - 256)] = (_Float16)(acc[r] + bias);
      } else {                           // v -> transposed [dim][token]
        v8h pk;
#pragma unroll
        for (int r = 0; r < 8; ++r) pk[r] = (_Float16)(acc[r] + bias);
        *(v8h*)&lvt[(col - 512) * VTS + mt * 16 + hi * 8] = pk;
      }
    }
  }
  __syncthreads();

  // ---- Stage 3: per-head attention (wave w == head h) ----
  {
    const int h = w;
    _Float16* lph = lp + h * NTOK * PSTR;

    // prefetch this wave's Wproj tiles (consumed in stage 4) into cache
#pragma unroll
    for (int jt = 0; jt < 2; ++jt)
      __builtin_prefetch(wproj_h + ((w * 2 + jt) * 16 + ln) * DIMC + hi * 128, 0, 3);

    // S = q k^T, scale, softmax, P -> LDS ; one query-row-block (mt) at a time
    // keeps only 4 accumulator tiles (32 VGPRs) live across the softmax.
#pragma unroll
    for (int mt = 0; mt < 4; ++mt) {
      v16h aq = load_a16x32(lq + (mt * 16) * XS + h * HDIM, XS);
      v8f s[4];
#pragma unroll
      for (int nt = 0; nt < 4; ++nt) {
        v8f z = {};
        v16h bk = load_b32x16(lk + (nt * 16) * XS + h * HDIM, XS);
        s[nt] = wmma32(aq, bk, z);
        s[nt] = s[nt] * SCALEF;
      }
#pragma unroll
      for (int r = 0; r < 8; ++r) {
        float mx = fmaxf(fmaxf(s[0][r], s[1][r]), fmaxf(s[2][r], s[3][r]));
        mx = fmaxf(mx, __shfl_xor(mx, 1));
        mx = fmaxf(mx, __shfl_xor(mx, 2));
        mx = fmaxf(mx, __shfl_xor(mx, 4));
        mx = fmaxf(mx, __shfl_xor(mx, 8));
        float sum = 0.f;
#pragma unroll
        for (int nt = 0; nt < 4; ++nt) {
          float e = __expf(s[nt][r] - mx);
          s[nt][r] = e;
          sum += e;
        }
        sum += __shfl_xor(sum, 1);
        sum += __shfl_xor(sum, 2);
        sum += __shfl_xor(sum, 4);
        sum += __shfl_xor(sum, 8);
        const float inv = 1.0f / sum;
#pragma unroll
        for (int nt = 0; nt < 4; ++nt)
          lph[(mt * 16 + hi * 8 + r) * PSTR + nt * 16 + ln] =
              (_Float16)(s[nt][r] * inv);
      }
    }

    // out_h = P @ V   (M=64, N=32, K=64); same-wave LDS RAW is kept in order.
#pragma unroll
    for (int mt = 0; mt < 4; ++mt) {
      v16h pa[2];
#pragma unroll
      for (int kt = 0; kt < 2; ++kt)
        pa[kt] = load_a16x32(lph + (mt * 16) * PSTR + kt * 32, PSTR);
#pragma unroll
      for (int ntl = 0; ntl < 2; ++ntl) {
        v8f o = {};
#pragma unroll
        for (int kt = 0; kt < 2; ++kt) {
          v16h bv = load_b32x16(lvt + (h * HDIM + ntl * 16) * VTS + kt * 32, VTS);
          o = wmma32(pa[kt], bv, o);
        }
#pragma unroll
        for (int r = 0; r < 8; ++r)
          lo[(mt * 16 + hi * 8 + r) * XS + h * HDIM + ntl * 16 + ln] = (_Float16)o[r];
      }
    }
  }
  __syncthreads();

  // ---- Stage 4: projection + scatter back to NCHW fp32 ----
  // wave w owns n-tiles [2w, 2w+2); M=64, N=256, K=256
  for (int mt = 0; mt < 4; ++mt) {
    v16h af[8];
#pragma unroll
    for (int ks = 0; ks < 8; ++ks)
      af[ks] = load_a16x32(lo + (mt * 16) * XS + ks * 32, XS);
    for (int jt = 0; jt < 2; ++jt) {
      const int nt = w * 2 + jt;
      v8f acc = {};
#pragma unroll
      for (int ks = 0; ks < 8; ++ks) {
        v16h bf = load_b32x16(wproj_h + (nt * 16) * DIMC + ks * 32, DIMC);
        acc = wmma32(af[ks], bf, acc);
      }
      const int ch = nt * 16 + ln;
      const float bias = bproj[ch];
      const int row = hw * 8 + mt * 2 + hi;    // pixel row of tokens in this D half
#pragma unroll
      for (int r = 0; r < 8; ++r)
        out[((b * DIMC + ch) * 256 + row) * 256 + ww * 8 + r] = acc[r] + bias;
    }
  }
}

extern "C" void kernel_launch(void* const* d_in, const int* in_sizes, int n_in,
                              void* d_out, int out_size, void* d_ws, size_t ws_size,
                              hipStream_t stream) {
  const float* x     = (const float*)d_in[0];
  const float* Wqkv  = (const float*)d_in[1];
  const float* bqkv  = (const float*)d_in[2];
  const float* Wproj = (const float*)d_in[3];
  const float* bproj = (const float*)d_in[4];
  float* out = (float*)d_out;

  _Float16* wqkv_h  = (_Float16*)d_ws;                                // 3*256*256 f16
  _Float16* wproj_h = (_Float16*)((char*)d_ws + 3 * DIMC * DIMC * 2); // 256*256 f16

  wa_convert_weights<<<768, 256, 0, stream>>>(Wqkv, Wproj, wqkv_h, wproj_h);

  const size_t lds_bytes = (size_t)LDS_HALVES * sizeof(_Float16);     // 245760 B
  wa_fused<<<4096, 256, lds_bytes, stream>>>(x, bqkv, bproj, wqkv_h, wproj_h, out);
}
